// TGCN_54271206752561
// MI455X (gfx1250) — matrix operand order repaired and checked by reading, not compiled, not run on hardware
//
#include <hip/hip_runtime.h>
#include <stdint.h>

// ---------------------------------------------------------------------------
// T-GCN for MI455X (gfx1250): bf16 WMMA GEMMs (double-buffered async LDS
// staging) + persistent-WG GRU recurrence
// ---------------------------------------------------------------------------

typedef __attribute__((ext_vector_type(16))) __bf16 v16bf;
typedef __attribute__((ext_vector_type(8)))  __bf16 v8bf;
typedef __attribute__((ext_vector_type(8)))  float  v8f;
typedef __attribute__((ext_vector_type(4)))  float  v4f;
typedef __attribute__((ext_vector_type(4)))  int    v4i;

#define SEQ   128
#define BATCH 32
#define NN    1024
#define FIN   16
#define NHID  2
#define GFEAT 4
#define HH    256
#define IN0   4096   // GFEAT*NN
#define H3    768    // 3*HH

#ifndef __has_builtin
#define __has_builtin(x) 0
#endif
#if __has_builtin(__builtin_amdgcn_global_load_async_to_lds_b128)
#define HAVE_ASYNC_LDS 1
#else
#define HAVE_ASYNC_LDS 0
#endif

// ------------------- async global->LDS 16B copy helpers --------------------
// builtin signature (from hipcc diagnostic): (AS1 v4i* src, AS3 v4i* dst, Ii, Ii)
static __device__ inline void async_copy_b128(const __bf16* __restrict__ g,
                                              __bf16* __restrict__ l) {
#if HAVE_ASYNC_LDS
    __attribute__((address_space(1))) v4i* gp =
        reinterpret_cast<__attribute__((address_space(1))) v4i*>(
            reinterpret_cast<uintptr_t>(g));
    __attribute__((address_space(3))) v4i* lp =
        reinterpret_cast<__attribute__((address_space(3))) v4i*>(
            static_cast<uint32_t>(reinterpret_cast<uintptr_t>(l)));
    __builtin_amdgcn_global_load_async_to_lds_b128(gp, lp, 0, 0);
#else
    *(v8bf*)l = *(const v8bf*)g;   // synchronous fallback
#endif
}

static __device__ inline void wait_async_lds() {
#if HAVE_ASYNC_LDS
#if __has_builtin(__builtin_amdgcn_s_wait_asynccnt)
    __builtin_amdgcn_s_wait_asynccnt(0);
#else
    asm volatile("s_wait_asynccnt 0x0" ::: "memory");
#endif
#endif
}

// ---------------- WMMA fragment loaders (ISA 7.12.2 layouts) ---------------
// A (16x32 bf16): lane L<16 -> row M=L, K = {0..7, 16..23}; lane>=16: K+8
static __device__ inline v16bf load_frag_a(const __bf16* __restrict__ row, int half) {
    v8bf c0 = *(const v8bf*)(row + 8 * half);        // K = 8*half + 0..7
    v8bf c1 = *(const v8bf*)(row + 16 + 8 * half);   // K = 16 + 8*half + 0..7
    v16bf r;
#pragma unroll
    for (int i = 0; i < 8; ++i) { r[i] = c0[i]; r[i + 8] = c1[i]; }
    return r;
}
// B (32x16 bf16), staged col-major [n][k]: lane col n=lane&15, K = {0..7, 16..23} + 8*half
static __device__ inline v16bf load_frag_b(const __bf16* __restrict__ row, int half) {
    v8bf c0 = *(const v8bf*)(row + 8 * half);
    v8bf c1 = *(const v8bf*)(row + 16 + 8 * half);
    v16bf r;
#pragma unroll
    for (int i = 0; i < 8; ++i) { r[i] = c0[i]; r[i + 8] = c1[i]; }
    return r;
}

// ------------------------------ f32 -> bf16 --------------------------------
__global__ void cvt_bf16_kernel(const float* __restrict__ src,
                                __bf16* __restrict__ dst, long long n) {
    long long i = (long long)blockIdx.x * blockDim.x + threadIdx.x;
    long long stride = (long long)gridDim.x * blockDim.x;
    for (; i < n; i += stride) dst[i] = (__bf16)src[i];
}

// ------------- K1: T1[s][b*2+h][n] = x[s,b,n,:] @ W1 + b1  (bf16) ----------
__global__ void gc1_kernel(const float* __restrict__ x,
                           const float* __restrict__ W1,
                           const float* __restrict__ b1,
                           __bf16* __restrict__ T1) {
    __shared__ float w[FIN * NHID];
    __shared__ float bb[NHID];
    if (threadIdx.x < FIN * NHID) w[threadIdx.x] = W1[threadIdx.x];
    if (threadIdx.x < NHID) bb[threadIdx.x] = b1[threadIdx.x];
    __syncthreads();
    long long idx = (long long)blockIdx.x * 256 + threadIdx.x; // (s*32+b)*1024+n
    int n = (int)(idx & 1023);
    int b = (int)((idx >> 10) & 31);
    int s = (int)(idx >> 15);
    const float* xp = x + (idx << 4);
    float t0 = bb[0], t1 = bb[1];
#pragma unroll
    for (int f = 0; f < FIN; ++f) {
        float xv = xp[f];
        t0 += xv * w[f * 2 + 0];
        t1 += xv * w[f * 2 + 1];
    }
    __bf16* o = T1 + ((long long)s * 64 + b * 2) * NN + n;
    o[0]  = (__bf16)t0;
    o[NN] = (__bf16)t1;
}

// ------- K3: T2[s][b*4+g][n] = relu'd H1[s][b*2+:][n] @ W2 + b2 (bf16) -----
__global__ void gc2_kernel(const float* __restrict__ W2,
                           const float* __restrict__ b2,
                           const float* __restrict__ H1,
                           __bf16* __restrict__ T2) {
    __shared__ float w[NHID * GFEAT];
    __shared__ float bb[GFEAT];
    if (threadIdx.x < NHID * GFEAT) w[threadIdx.x] = W2[threadIdx.x];
    if (threadIdx.x < GFEAT) bb[threadIdx.x] = b2[threadIdx.x];
    __syncthreads();
    long long idx = (long long)blockIdx.x * 256 + threadIdx.x;
    int n = (int)(idx & 1023);
    int b = (int)((idx >> 10) & 31);
    int s = (int)(idx >> 15);
    const float* hp = H1 + ((long long)s * 64 + b * 2) * NN + n;
    float h0 = hp[0], h1v = hp[NN];
    __bf16* o = T2 + ((long long)s * 128 + b * 4) * NN + n;
#pragma unroll
    for (int g = 0; g < GFEAT; ++g) {
        float t = bb[g] + h0 * w[g] + h1v * w[GFEAT + g];
        o[(long long)g * NN] = (__bf16)t;
    }
}

// ----------------------- batched WMMA GEMM (bf16->f32) ---------------------
// C[m,col] = sum_k A[m,k] * Bm[col,k]   (A row-major MxK, Bm "col-major" NxK)
// colMajor=1: C stored [col][m] stride M (+relu). colMajor=0: [m][col] (+bias)
// Double-buffered LDS; K-tile k+1 streams in via GLOBAL_LOAD_ASYNC_TO_LDS
// while WMMA consumes tile k; drained with s_wait_asynccnt.
#define BM 128
#define BN 64
#define BK 32
#define LDT 40   // padded LDS stride (bank-conflict-free, 16B aligned rows)

__global__ __launch_bounds__(256, 2) void wmma_gemm_kernel(
    const __bf16* __restrict__ A, const __bf16* __restrict__ Bmat,
    float* __restrict__ C, const float* __restrict__ bias,
    int M, int N, int K,
    long long sA, long long sB, long long sC,
    int relu, int colMajor) {
    __shared__ __bf16 As[2][BM * LDT];
    __shared__ __bf16 Bs[2][BN * LDT];
    int batch = blockIdx.y;
    A    += (long long)batch * sA;
    Bmat += (long long)batch * sB;
    C    += (long long)batch * sC;
    int mtiles = M / BM;
    int mT = blockIdx.x % mtiles;
    int nT = blockIdx.x / mtiles;
    int tid = threadIdx.x;
    int wave = tid >> 5, lane = tid & 31, half = lane >> 4, ln = lane & 15;
    int wm = wave & 3, wn = wave >> 2;       // 4x2 wave grid, 32x32 per wave
    v8f acc[2][2] = {};

    // stage one K-tile (16B async chunks; LDS rows padded to LDT)
    auto stage = [&](int bsel, int k0) {
        for (int i = tid; i < BM * 4; i += 256) {        // 128 rows x 4 chunks
            int r = i >> 2, c = (i & 3) * 8;
            async_copy_b128(A + (long long)(mT * BM + r) * K + k0 + c,
                            &As[bsel][r * LDT + c]);
        }
        for (int i = tid; i < BN * 4; i += 256) {        // 64 cols x 4 chunks
            int r = i >> 2, c = (i & 3) * 8;
            async_copy_b128(Bmat + (long long)(nT * BN + r) * K + k0 + c,
                            &Bs[bsel][r * LDT + c]);
        }
    };

    stage(0, 0);
    wait_async_lds();
    __syncthreads();
    int buf = 0;

    for (int k0 = 0; k0 < K; k0 += BK) {
        if (k0 + BK < K) stage(buf ^ 1, k0 + BK);   // prefetch next tile

        v16bf af[2], bfb[2];
#pragma unroll
        for (int fm = 0; fm < 2; ++fm)
            af[fm] = load_frag_a(&As[buf][(wm * 32 + fm * 16 + ln) * LDT], half);
#pragma unroll
        for (int fn = 0; fn < 2; ++fn)
            bfb[fn] = load_frag_b(&Bs[buf][(wn * 32 + fn * 16 + ln) * LDT], half);
#pragma unroll
        for (int fm = 0; fm < 2; ++fm)
#pragma unroll
            for (int fn = 0; fn < 2; ++fn)
                acc[fm][fn] = __builtin_amdgcn_wmma_f32_16x16x32_bf16(
                    false, af[fm], false, bfb[fn], (short)0, acc[fm][fn],
                    false, false);

        wait_async_lds();
        __syncthreads();
        buf ^= 1;
    }

#pragma unroll
    for (int fm = 0; fm < 2; ++fm)
#pragma unroll
        for (int fn = 0; fn < 2; ++fn) {
            int col = nT * BN + wn * 32 + fn * 16 + ln;
            int mb  = mT * BM + wm * 32 + fm * 16 + 8 * half; // 8 consecutive m
            v8f c = acc[fm][fn];
            if (relu) {
#pragma unroll
                for (int r = 0; r < 8; ++r) c[r] = fmaxf(c[r], 0.0f);
            }
            if (colMajor) {
                float* p = C + (long long)col * M + mb;
                v4f lo = {c[0], c[1], c[2], c[3]};
                v4f hi = {c[4], c[5], c[6], c[7]};
                *(v4f*)p = lo;
                *(v4f*)(p + 4) = hi;
            } else {
                float bv = bias ? bias[col] : 0.0f;
#pragma unroll
                for (int r = 0; r < 8; ++r)
                    C[(long long)(mb + r) * N + col] = c[r] + bv;
            }
        }
}

// --------- log_softmax over node axis; emit bf16 in GRU-input layout -------
// S3 layout [s][b*4+g][m] (f32). X0[(s*32+b)][n*4+g] = bf16(log_softmax)
__global__ void logsoftmax_kernel(const float* __restrict__ S3,
                                  __bf16* __restrict__ X0) {
    __shared__ float red[256];
    int blk = blockIdx.x;
    int s = blk >> 7, col = blk & 127, b = col >> 2, g = col & 3;
    const float* v = S3 + ((long long)s * 128 + col) * NN;
    int tid = threadIdx.x;

    float lmax = -1e30f;
    for (int i = tid; i < NN; i += 256) lmax = fmaxf(lmax, v[i]);
    red[tid] = lmax;
    __syncthreads();
    for (int sft = 128; sft > 0; sft >>= 1) {
        if (tid < sft) red[tid] = fmaxf(red[tid], red[tid + sft]);
        __syncthreads();
    }
    float mx = red[0];
    __syncthreads();

    float lsum = 0.0f;
    for (int i = tid; i < NN; i += 256) lsum += __expf(v[i] - mx);
    red[tid] = lsum;
    __syncthreads();
    for (int sft = 128; sft > 0; sft >>= 1) {
        if (tid < sft) red[tid] += red[tid + sft];
        __syncthreads();
    }
    float lse = __logf(red[0]);

    __bf16* o = X0 + ((long long)s * 32 + b) * IN0 + g;
    for (int i = tid; i < NN; i += 256)
        o[(long long)i * 4] = (__bf16)(v[i] - mx - lse);
}

// -------------- persistent single-WG GRU layer (WMMA recurrence) -----------
// GI: precomputed x@W_ih^T + b_ih, rows (t*32+b), 768 cols (r|z|n), f32
// Whh: bf16 [j][k] (768x256). Per step: gh = h@Whh^T + b_hh via WMMA.
__global__ __launch_bounds__(512, 1) void gru_layer_kernel(
    const float* __restrict__ GI, const __bf16* __restrict__ Whh,
    const float* __restrict__ bhh,
    __bf16* __restrict__ outBf, float* __restrict__ outF,
    float* __restrict__ hn) {
    __shared__ float  gh[32 * 776];      // 32 x 768 (+pad)
    __shared__ float  hf[32 * 256];      // h state f32
    __shared__ __bf16 hb[32 * 264];      // h state bf16 (+pad) for WMMA A

    int tid = threadIdx.x, wave = tid >> 5, lane = tid & 31;
    int half = lane >> 4, ln = lane & 15;

    for (int i = tid; i < 32 * 256; i += 512) hf[i] = 0.0f;
    for (int i = tid; i < 32 * 264; i += 512) hb[i] = (__bf16)0.0f;
    __syncthreads();

    for (int t = 0; t < SEQ; ++t) {
        // prefetch next timestep's input-gate rows into cache
        if (t + 1 < SEQ)
            __builtin_prefetch(GI + ((long long)(t + 1) * 32) * H3 +
                                   (tid & 127) * 192, 0, 3);

        // gh = h @ Whh^T + bhh : M=32 (2 tiles) x N=768 (48 tiles), K=256
        // 96 (m,n) tile pairs over 16 waves -> 6 each; Whh frags direct from L2
        for (int p = wave * 6; p < wave * 6 + 6; ++p) {
            int mt = p / 48, nt = p % 48;
            int j = nt * 16 + ln;
            const __bf16* wrow = Whh + (long long)j * HH;
            const __bf16* hrow = hb + (mt * 16 + ln) * 264;
            v8f acc = {};
#pragma unroll
            for (int ks = 0; ks < 8; ++ks) {
                v16bf a = load_frag_a(hrow + ks * 32, half);
                v16bf bfr = load_frag_b(wrow + ks * 32, half);
                acc = __builtin_amdgcn_wmma_f32_16x16x32_bf16(
                    false, a, false, bfr, (short)0, acc, false, false);
            }
            float bj = bhh[j];
            int brow = mt * 16 + 8 * half;
#pragma unroll
            for (int r = 0; r < 8; ++r) gh[(brow + r) * 776 + j] = acc[r] + bj;
        }
        __syncthreads();

        // gate math + state update (torch GRU: r,z,n ordering)
        for (int u = tid; u < 32 * 256; u += 512) {
            int b = u >> 8, q = u & 255;
            const float* gi = GI + ((long long)(t * 32 + b)) * H3 + q;
            float ir = gi[0], iz = gi[HH], in_ = gi[2 * HH];
            float hr = gh[b * 776 + q];
            float hz = gh[b * 776 + HH + q];
            float hnn = gh[b * 776 + 2 * HH + q];
            float r = 1.0f / (1.0f + __expf(-(ir + hr)));
            float z = 1.0f / (1.0f + __expf(-(iz + hz)));
            float n = tanhf(in_ + r * hnn);
            float hnew = (1.0f - z) * n + z * hf[b * 256 + q];
            hf[b * 256 + q] = hnew;
            hb[b * 264 + q] = (__bf16)hnew;
            long long orow = ((long long)t * 32 + b) * HH + q;
            if (outBf) outBf[orow] = (__bf16)hnew;
            if (outF)  outF[orow]  = hnew;
        }
        __syncthreads();
    }
    for (int i = tid; i < 32 * 256; i += 512) hn[i] = hf[i];
}

// --------------------------------- host ------------------------------------
extern "C" void kernel_launch(void* const* d_in, const int* in_sizes, int n_in,
                              void* d_out, int out_size, void* d_ws, size_t ws_size,
                              hipStream_t stream) {
    const float* x       = (const float*)d_in[0];
    const float* adj     = (const float*)d_in[1];
    const float* gc1_w   = (const float*)d_in[2];
    const float* gc1_b   = (const float*)d_in[3];
    const float* gc2_w   = (const float*)d_in[4];
    const float* gc2_b   = (const float*)d_in[5];
    const float* w_ih_l0 = (const float*)d_in[6];
    const float* w_hh_l0 = (const float*)d_in[7];
    const float* b_ih_l0 = (const float*)d_in[8];
    const float* b_hh_l0 = (const float*)d_in[9];
    const float* w_ih_l1 = (const float*)d_in[10];
    const float* w_hh_l1 = (const float*)d_in[11];
    const float* b_ih_l1 = (const float*)d_in[12];
    const float* b_hh_l1 = (const float*)d_in[13];

    char* ws = (char*)d_ws;
    size_t off = 0;
    auto take = [&](size_t bytes) { void* p = ws + off; off += (bytes + 255) & ~(size_t)255; return p; };

    __bf16* adj_bf  = (__bf16*)take((size_t)NN * NN * 2);
    __bf16* wih0_bf = (__bf16*)take((size_t)H3 * IN0 * 2);
    __bf16* wih1_bf = (__bf16*)take((size_t)H3 * HH * 2);
    __bf16* whh0_bf = (__bf16*)take((size_t)H3 * HH * 2);
    __bf16* whh1_bf = (__bf16*)take((size_t)H3 * HH * 2);
    __bf16* T1      = (__bf16*)take((size_t)SEQ * 64 * NN * 2);
    float*  H1      = (float*) take((size_t)SEQ * 64 * NN * 4);
    __bf16* T2      = (__bf16*)take((size_t)SEQ * 128 * NN * 2);
    float*  S3      = (float*) take((size_t)SEQ * 128 * NN * 4);
    __bf16* X0      = (__bf16*)take((size_t)SEQ * BATCH * IN0 * 2);
    float*  GI0     = (float*) take((size_t)SEQ * BATCH * H3 * 4);
    float*  GI1     = (float*) take((size_t)SEQ * BATCH * H3 * 4);
    __bf16* out0_bf = (__bf16*)take((size_t)SEQ * BATCH * HH * 2);
    (void)ws_size; (void)in_sizes; (void)n_in; (void)out_size;

    float* out_f = (float*)d_out;                       // (128,32,256)
    float* hn0   = out_f + (size_t)SEQ * BATCH * HH;    // hn[0]
    float* hn1   = hn0 + (size_t)BATCH * HH;            // hn[1]

    // weight / adjacency conversions to bf16
    cvt_bf16_kernel<<<2048, 256, 0, stream>>>(adj,     adj_bf,  (long long)NN * NN);
    cvt_bf16_kernel<<<2048, 256, 0, stream>>>(w_ih_l0, wih0_bf, (long long)H3 * IN0);
    cvt_bf16_kernel<<<512,  256, 0, stream>>>(w_ih_l1, wih1_bf, (long long)H3 * HH);
    cvt_bf16_kernel<<<512,  256, 0, stream>>>(w_hh_l0, whh0_bf, (long long)H3 * HH);
    cvt_bf16_kernel<<<512,  256, 0, stream>>>(w_hh_l1, whh1_bf, (long long)H3 * HH);

    // GCN layer 1 feature transform, then adj GEMM (+relu)
    gc1_kernel<<<16384, 256, 0, stream>>>(x, gc1_w, gc1_b, T1);
    wmma_gemm_kernel<<<dim3(8, SEQ), 256, 0, stream>>>(
        adj_bf, T1, H1, nullptr, NN, 64, NN,
        0LL, 64LL * NN, 64LL * NN, /*relu=*/1, /*colMajor=*/1);

    // GCN layer 2 feature transform, then adj GEMM
    gc2_kernel<<<16384, 256, 0, stream>>>(gc2_w, gc2_b, H1, T2);
    wmma_gemm_kernel<<<dim3(16, SEQ), 256, 0, stream>>>(
        adj_bf, T2, S3, nullptr, NN, 128, NN,
        0LL, 128LL * NN, 128LL * NN, /*relu=*/0, /*colMajor=*/1);

    // log_softmax over nodes -> bf16 GRU input X0 (4096 x 4096)
    logsoftmax_kernel<<<SEQ * 128, 256, 0, stream>>>(S3, X0);

    // GRU layer 0: batched input GEMM (4096x768x4096), then recurrence
    wmma_gemm_kernel<<<dim3(32 * 12, 1), 256, 0, stream>>>(
        X0, wih0_bf, GI0, b_ih_l0, SEQ * BATCH, H3, IN0,
        0LL, 0LL, 0LL, /*relu=*/0, /*colMajor=*/0);
    gru_layer_kernel<<<1, 512, 0, stream>>>(GI0, whh0_bf, b_hh_l0,
                                            out0_bf, nullptr, hn0);

    // GRU layer 1: input GEMM (4096x768x256), then recurrence -> d_out
    wmma_gemm_kernel<<<dim3(32 * 12, 1), 256, 0, stream>>>(
        out0_bf, wih1_bf, GI1, b_ih_l1, SEQ * BATCH, H3, HH,
        0LL, 0LL, 0LL, /*relu=*/0, /*colMajor=*/0);
    gru_layer_kernel<<<1, 512, 0, stream>>>(GI1, whh1_bf, b_hh_l1,
                                            nullptr, out_f, hn1);
}
